// NewGATCell_47605417509430
// MI455X (gfx1250) — matrix operand order repaired and checked by reading, not compiled, not run on hardware
//
#include <hip/hip_runtime.h>

// NewGATCell fused kernel for MI455X (gfx1250, wave32).
// One 256-thread (8-wave) workgroup per (b,s) tile; both GEMMs via
// v_wmma_f32_16x16x32_bf16 with f32 accumulation; ht / logits / softmax live
// entirely in LDS so HBM traffic stays ~80MB (h + adj + out) => memory-bound
// at 23.3 TB/s (~3.5us); compute ~6.7 GFLOP fits easily in WMMA throughput.

#define LEAKY  0.02f
#define NEGF  -9000000000000000.0f

typedef __attribute__((ext_vector_type(16))) __bf16       v16bf;
typedef __attribute__((ext_vector_type(2)))  __bf16       v2bf;
typedef __attribute__((ext_vector_type(8)))  float        v8f;
typedef __attribute__((ext_vector_type(4)))  float        v4f;
typedef __attribute__((ext_vector_type(2)))  float        v2f;
typedef __attribute__((ext_vector_type(8)))  unsigned int v8u;
typedef __attribute__((ext_vector_type(4)))  unsigned int v4u;
typedef __attribute__((ext_vector_type(2)))  unsigned int v2u;

// Pack two f32 into one dword of bf16 via a single v_cvt_pk_bf16_f32.
__device__ __forceinline__ unsigned int pack2(float lo, float hi) {
  v2bf p = __builtin_convertvector((v2f){lo, hi}, v2bf);
  return __builtin_bit_cast(unsigned int, p);
}
__device__ __forceinline__ unsigned short bfc(float x) {
  __bf16 b = (__bf16)x;
  return __builtin_bit_cast(unsigned short, b);
}
__device__ __forceinline__ float bf2f(unsigned short s) {
  return __uint_as_float(((unsigned int)s) << 16);
}

// A-matrix 16x32 bf16 fragment: lane m = l&15,
// lanes 0-15 hold K 0..15, lanes 16-31 hold K 16..31,
// two K-values packed per VGPR. Source: row-major [m][k] LDS tile.
__device__ __forceinline__ v16bf load_fragA(const unsigned short* mat, int row0,
                                            int k0, int stride) {
  const int lane = threadIdx.x & 31;
  const unsigned short* p = mat + (row0 + (lane & 15)) * stride + k0 + ((lane >> 4) << 4);
  v8u raw;
#pragma unroll
  for (int v = 0; v < 8; ++v) {
    const int k = 2 * v;   // {0,2,4,...,14}
    raw[v] = *(const unsigned int*)(p + k);
  }
  return __builtin_bit_cast(v16bf, raw);
}

// B-matrix 32x16 bf16 fragment: lane n = l&15, lanes 0-15 hold K 0..15,
// lanes 16-31 hold K 16..31, packed pairs. Source: column-major-stored
// LDS tile [n][k] -> 16 contiguous ushorts per lane.
__device__ __forceinline__ v16bf load_fragB(const unsigned short* mat, int row0,
                                            int k0, int stride) {
  const int lane = threadIdx.x & 31;
  const unsigned short* p = mat + (row0 + (lane & 15)) * stride + k0 + ((lane >> 4) << 4);
  v8u raw;
#pragma unroll
  for (int v = 0; v < 8; ++v) raw[v] = *(const unsigned int*)(p + 2 * v);
  return __builtin_bit_cast(v16bf, raw);
}

// ---------------------------------------------------------------- prep kernels

// Pack W (64x256 f32, row-major) -> W^T bf16 [c][k] (256x64) in workspace.
__global__ __launch_bounds__(256) void prep_wt(const float* __restrict__ W,
                                               unsigned short* __restrict__ wt) {
  const int i = blockIdx.x * 256 + threadIdx.x;   // i = k*256 + c
  const int k = i >> 8, c = i & 255;
  wt[c * 64 + k] = bfc(W[i]);
}

// m[b,i,k] = mean over s of adj[b,s,i,k]   (vectorized: one float4 per thread)
__global__ __launch_bounds__(256) void mean_adj(const float* __restrict__ adj,
                                                float* __restrict__ m) {
  const int idx4 = blockIdx.x * 256 + threadIdx.x;   // < 32*1024
  const int b = idx4 >> 10, rest = idx4 & 1023;
  const float* p = adj + (size_t)b * 50 * 4096 + rest * 4;
  v4f s = {0.f, 0.f, 0.f, 0.f};
#pragma unroll 10
  for (int t = 0; t < 50; ++t) s += *(const v4f*)(p + t * 4096);
  *(v4f*)(m + idx4 * 4) = s * (1.f / 50.f);
}

// thr[b,o] = sum_{i,k} sp_w[o,i,k] * m[b,i,k] + sp_b[o]
__global__ __launch_bounds__(256) void thr_kernel(const float* __restrict__ mmat,
                                                  const float* __restrict__ spw,
                                                  const float* __restrict__ spb,
                                                  float* __restrict__ thr) {
  __shared__ __align__(16) float mL[4096];
  __shared__ float red[256];
  const int b = blockIdx.x, t = threadIdx.x;
  const float* mb = mmat + b * 4096;
#pragma unroll
  for (int q = 0; q < 4; ++q) {
    const int i4 = t + q * 256;
    *(v4f*)&mL[i4 * 4] = *(const v4f*)(mb + i4 * 4);
  }
  __syncthreads();
  for (int o = 0; o < 64; ++o) {
    const float* w = spw + o * 4096 + t * 16;
    v4f acc = {0.f, 0.f, 0.f, 0.f};
#pragma unroll
    for (int q = 0; q < 4; ++q)
      acc += *(const v4f*)(w + q * 4) * *(const v4f*)&mL[t * 16 + q * 4];
    red[t] = acc[0] + acc[1] + acc[2] + acc[3];
    __syncthreads();
    for (int off = 128; off > 0; off >>= 1) {
      if (t < off) red[t] += red[t + off];
      __syncthreads();
    }
    if (t == 0) thr[b * 64 + o] = red[0] + spb[o];
    __syncthreads();
  }
}

// ---------------------------------------------------------------- fused kernel

struct __align__(16) Smem {
  union {
    struct { unsigned short hA[64 * 68]; unsigned short wB[256 * 68]; } g1;  // GEMM1 staging
    unsigned short a2[64 * 264];                                             // softmax out (A of GEMM2)
  } u;
  unsigned short bt2[64 * 264];   // ht as [d][h*64+j] bf16 (B of GEMM2, col-stored)
  float adjT[64 * 68];            // adj tile, padded stride (16B-aligned rows)
  float srcv[256];                // src[j,h]
  float dstv[256];                // dst[i,h]
  float thrv[64];                 // thr[b, i]
};

__global__ __launch_bounds__(256) void gat_fused(const float* __restrict__ h,
                                                 const float* __restrict__ adj,
                                                 const unsigned short* __restrict__ wt,
                                                 const float* __restrict__ a_w,
                                                 const float* __restrict__ thr,
                                                 float* __restrict__ out) {
  __shared__ Smem sm;
  const int bs   = blockIdx.x;         // b*S + s
  const int b    = bs / 50;
  const int tid  = threadIdx.x;
  const int lane = tid & 31;
  const int wv   = tid >> 5;           // 8 waves
  const int n_l  = lane & 15;
  const int mb   = (lane >> 4) << 3;

  const float* hb = h   + (size_t)bs * 4096;
  const float* ab = adj + (size_t)bs * 4096;
  __builtin_prefetch(ab, 0, 1);        // global_prefetch_b8

  // Stage with b128 loads: h tile -> bf16 [n][f] (pad 68); adj tile f32 (pad 68);
  // W^T bf16 [n][k] (pad 68); thr row.
#pragma unroll
  for (int q = 0; q < 4; ++q) {
    const int i4 = tid + q * 256;                 // 0..1023 float4s
    const v4f hv = *(const v4f*)(hb + i4 * 4);
    const v4f av = *(const v4f*)(ab + i4 * 4);
    const int n = i4 >> 4, f = (i4 & 15) << 2;
    *(v2u*)&sm.u.g1.hA[n * 68 + f] = (v2u){pack2(hv[0], hv[1]), pack2(hv[2], hv[3])};
    *(v4f*)&sm.adjT[n * 68 + f] = av;
  }
#pragma unroll
  for (int q = 0; q < 8; ++q) {
    const int i4 = tid + q * 256;                 // 0..2047 (8 bf16 each)
    const v4u w = *(const v4u*)(wt + i4 * 8);
    const int n = i4 >> 3, k = (i4 & 7) << 3;
    *(v2u*)&sm.u.g1.wB[n * 68 + k]     = (v2u){w[0], w[1]};
    *(v2u*)&sm.u.g1.wB[n * 68 + k + 4] = (v2u){w[2], w[3]};
  }
  if (tid < 64) sm.thrv[tid] = thr[b * 64 + tid];
  __syncthreads();

  // GEMM1: ht(64x256) = h(64x64) @ W(64x256); epilogue scatters bf16 pairs into
  // bt2[d][hh*64+j] so it is directly the column-stored B operand of GEMM2.
#pragma unroll
  for (int ci = 0; ci < 2; ++ci) {
    const int ct = wv * 2 + ci;        // column tile 0..15
#pragma unroll
    for (int mt = 0; mt < 4; ++mt) {   // row tile
      v8f acc = {0.f, 0.f, 0.f, 0.f, 0.f, 0.f, 0.f, 0.f};
#pragma unroll
      for (int ks = 0; ks < 2; ++ks) {
        v16bf av = load_fragA(sm.u.g1.hA, mt * 16, ks * 32, 68);
        v16bf bv = load_fragB(sm.u.g1.wB, ct * 16, ks * 32, 68);
        acc = __builtin_amdgcn_wmma_f32_16x16x32_bf16(false, av, false, bv,
                                                      (short)0, acc, false, false);
      }
      const int c = ct * 16 + n_l;
      const int hh = c >> 6, dd = c & 63;
#pragma unroll
      for (int r = 0; r < 8; r += 2) {           // consecutive j -> one b32 store
        const int j = mt * 16 + mb + r;
        *(unsigned int*)&sm.bt2[dd * 264 + hh * 64 + j] = pack2(acc[r], acc[r + 1]);
      }
    }
  }
  __syncthreads();

  // src[j,h] = ht[j,h,:]·a_src ; dst[j,h] = ht[j,h,:]·a_dst  (one thread per (j,h))
  {
    const int j = tid >> 2, hh = tid & 3;
    float s0 = 0.f, s1 = 0.f;
#pragma unroll 8
    for (int d = 0; d < 64; ++d) {
      const float v = bf2f(sm.bt2[d * 264 + hh * 64 + j]);
      s0 += v * a_w[d];
      s1 += v * a_w[64 + d];
    }
    sm.srcv[tid] = s0;   // [j*4+h]
    sm.dstv[tid] = s1;
  }
  __syncthreads();

  // Masked leaky-relu softmax over j; write a2[i][hh*64+j] bf16 pairs.
  // (a2 aliases the dead GEMM1 staging union.)
  {
    const int i = tid >> 2, hh = tid & 3;
    const float ti = sm.thrv[i];
    const float dv = sm.dstv[i * 4 + hh];
    float mx = -3.4e38f;
#pragma unroll 4
    for (int j = 0; j < 64; ++j) {
      float x = dv + sm.srcv[j * 4 + hh];
      x = (x >= 0.f) ? x : LEAKY * x;
      if (sm.adjT[i * 68 + j] < ti) x = NEGF;
      mx = fmaxf(mx, x);
    }
    float sum = 0.f;
#pragma unroll 4
    for (int j = 0; j < 64; ++j) {
      float x = dv + sm.srcv[j * 4 + hh];
      x = (x >= 0.f) ? x : LEAKY * x;
      if (sm.adjT[i * 68 + j] < ti) x = NEGF;
      sum += __expf(x - mx);
    }
    const float rinv = __builtin_amdgcn_rcpf(sum);
#pragma unroll 2
    for (int j = 0; j < 64; j += 2) {
      float x0 = dv + sm.srcv[j * 4 + hh];
      x0 = (x0 >= 0.f) ? x0 : LEAKY * x0;
      if (sm.adjT[i * 68 + j] < ti) x0 = NEGF;
      float x1 = dv + sm.srcv[(j + 1) * 4 + hh];
      x1 = (x1 >= 0.f) ? x1 : LEAKY * x1;
      if (sm.adjT[i * 68 + j + 1] < ti) x1 = NEGF;
      *(unsigned int*)&sm.u.a2[i * 264 + hh * 64 + j] =
          pack2(__expf(x0 - mx) * rinv, __expf(x1 - mx) * rinv);
    }
  }
  __syncthreads();

  // GEMM2 (+ head mean): out(64x64) = 0.25 * a2(64x256) @ bt2^T(256x64), K = h*64+j.
  float* ob = out + (size_t)bs * 4096;
#pragma unroll
  for (int ci = 0; ci < 2; ++ci) {
    const int t  = wv * 2 + ci;        // output tile 0..15
    const int mt = t >> 2, nt = t & 3;
    v8f acc = {0.f, 0.f, 0.f, 0.f, 0.f, 0.f, 0.f, 0.f};
#pragma unroll
    for (int ks = 0; ks < 8; ++ks) {
      v16bf av = load_fragA(sm.u.a2, mt * 16, ks * 32, 264);
      v16bf bv = load_fragB(sm.bt2,  nt * 16, ks * 32, 264);
      acc = __builtin_amdgcn_wmma_f32_16x16x32_bf16(false, av, false, bv,
                                                    (short)0, acc, false, false);
    }
    const int d = nt * 16 + n_l;
#pragma unroll
    for (int r = 0; r < 8; ++r) {
      const int i = mt * 16 + mb + r;
      ob[i * 64 + d] = 0.25f * acc[r];
    }
  }
}

// ---------------------------------------------------------------- launch

extern "C" void kernel_launch(void* const* d_in, const int* in_sizes, int n_in,
                              void* d_out, int out_size, void* d_ws, size_t ws_size,
                              hipStream_t stream) {
  const float* h   = (const float*)d_in[0];
  const float* adj = (const float*)d_in[1];
  const float* W   = (const float*)d_in[2];
  const float* a_w = (const float*)d_in[3];
  const float* spw = (const float*)d_in[4];
  const float* spb = (const float*)d_in[5];
  float* out = (float*)d_out;

  char* ws = (char*)d_ws;
  unsigned short* wt = (unsigned short*)ws;                 // 32 KB: W^T bf16
  float* m   = (float*)(ws + 32768);                        // 512 KB: adj mean
  float* thr = (float*)(ws + 32768 + 131072 * 4);           // 8 KB: thresholds

  prep_wt  <<<64,   256, 0, stream>>>(W, wt);
  mean_adj <<<128,  256, 0, stream>>>(adj, m);
  thr_kernel<<<32,  256, 0, stream>>>(m, spw, spb, thr);
  gat_fused<<<1600, 256, 0, stream>>>(h, adj, wt, a_w, thr, out);
}
